// T5Gemma2EncoderAttention_39582418599956
// MI455X (gfx1250) — compile-verified, matrix operand order checked by reading, not run on hardware
//
#include <hip/hip_runtime.h>
#include <hip/hip_bf16.h>

// ---------------------------------------------------------------------------
// T5Gemma2 encoder attention for MI455X (gfx1250, wave32, WMMA).
// bf16 operands + fp32 accumulation via v_wmma_f32_16x16x32_bf16 for all
// matmuls; fp32 VALU for RMS-norm / RoPE / softmax.
// Round 5: __launch_bounds__(.,1) on the WMMA kernels -> full register
// budget, no scratch spills in the hot loops (occupancy trade is correct for
// a matrix-pipe-bound kernel with software pipelining).
// ---------------------------------------------------------------------------

typedef __attribute__((ext_vector_type(16))) __bf16 v16bf;
typedef __attribute__((ext_vector_type(8)))  __bf16 v8bf;
typedef __attribute__((ext_vector_type(8)))  float  v8f;

#define B_    2
#define S_    2048
#define HID_  2048
#define NH_   16
#define NKV_  8
#define HD_   256
#define QKVF_ 8192                 // (NH + 2*NKV) * HD
#define SCALE_ 0.0625f             // 256^-0.5
#define EPS_   1e-6f

// ---- WMMA helpers ----------------------------------------------------------

__device__ __forceinline__ v8f wmma_bf16(v16bf a, v16bf b, v8f c) {
  // (neg_a, A, neg_b, B, c_mod, C, reuse_a, reuse_b)
  return __builtin_amdgcn_wmma_f32_16x16x32_bf16(false, a, false, b,
                                                 (short)0, c, false, false);
}

// Load a 16x32 bf16 A-style fragment (also for B via B^T) given a UNIFORM
// base pointer and a per-lane 32-bit element offset.  Per ISA 7.12.2:
//   lanes 0-15 : row = row0+lane,    K chunks {0..7} and {16..23}
//   lanes16-31 : row = row0+lane-16, K chunks {8..15} and {24..31}
__device__ __forceinline__ v16bf load_frag_off(const __bf16* base, unsigned off) {
  v8bf lo = *(const v8bf*)(base + off);
  v8bf hi = *(const v8bf*)(base + off + 16);
  v16bf out;
#pragma unroll
  for (int i = 0; i < 8; ++i) { out[i] = lo[i]; out[i + 8] = hi[i]; }
  return out;
}

// per-lane element offset for a fragment in a row-major [rows][ld] matrix
__device__ __forceinline__ unsigned frag_off(int row0, int ld, int lane) {
  return (unsigned)(row0 + (lane & 15)) * (unsigned)ld + (unsigned)((lane >> 4) * 8);
}

// generic-pointer version (used for LDS-staged fragments)
__device__ __forceinline__ v16bf load_frag_lds(const __bf16* base, int ld,
                                               int k0, int lane) {
  const __bf16* p = base + (unsigned)(lane & 15) * ld + k0 + (lane >> 4) * 8;
  v8bf lo = *(const v8bf*)(p);
  v8bf hi = *(const v8bf*)(p + 16);
  v16bf out;
#pragma unroll
  for (int i = 0; i < 8; ++i) { out[i] = lo[i]; out[i + 8] = hi[i]; }
  return out;
}

// ---- prep kernels ----------------------------------------------------------

__global__ void cvt_f32_to_bf16(const float* __restrict__ in,
                                __bf16* __restrict__ out, int n) {
  int i = blockIdx.x * blockDim.x + threadIdx.x;
  if (i < n) out[i] = (__bf16)in[i];
}

// out[c][r] = (bf16) in[r][c]   (produces B^T layout for WMMA B fragments)
__global__ void transpose_f32_to_bf16(const float* __restrict__ in,
                                      __bf16* __restrict__ out, int R, int C) {
  int i = blockIdx.x * blockDim.x + threadIdx.x;
  if (i < R * C) {
    int r = i / C, c = i - r * C;
    out[(size_t)c * R + r] = (__bf16)in[i];
  }
}

// ---- generic bf16 GEMM: each wave computes a 16(M) x 64(N) tile ------------
// A  : [M][K] bf16 row-major,  Bt : [N][K] bf16 row-major (i.e. B^T)
// Output: bf16 (Cb) or fp32 (Cf), row-major [M][N].
// K must be a multiple of 64 (call sites use 2048 / 4096).
// Double-buffered fragments; one per-lane offset per operand stream.

__global__ void __launch_bounds__(256, 1)
gemm16x64_bf16(const __bf16* __restrict__ A,
               const __bf16* __restrict__ Bt,
               __bf16* __restrict__ Cb,
               float*  __restrict__ Cf,
               int M, int N, int K) {
  const int lane = threadIdx.x & 31;
  const int wave = threadIdx.x >> 5;
  const int nt64 = N >> 6;
  const int gid  = blockIdx.x * 8 + wave;
  const int im   = gid / nt64;
  const int in64 = gid - im * nt64;
  if (im * 16 >= M) return;
  const int m0 = im * 16;
  const int n0 = in64 * 64;

  // loop-invariant per-lane offsets (elements) -- one per stream
  const unsigned offA = frag_off(m0, K, lane);
  const unsigned offB = frag_off(n0, K, lane);
  const int tstep = 16 * K;                 // uniform B-tile stride (elements)

  v8f acc[4] = {};
  // ping-pong fragment buffers
  v16bf aP, aQ;
  v16bf bP[4], bQ[4];
  aP = load_frag_off(A, offA);
#pragma unroll
  for (int t = 0; t < 4; ++t) bP[t] = load_frag_off(Bt + t * tstep, offB);

  for (int k0 = 0; k0 < K; k0 += 64) {
    // issue loads for k0+32 (buffer Q) before consuming buffer P
    aQ = load_frag_off(A + k0 + 32, offA);
#pragma unroll
    for (int t = 0; t < 4; ++t)
      bQ[t] = load_frag_off(Bt + t * tstep + k0 + 32, offB);
    __builtin_prefetch(A + k0 + 128 + offA, 0, 1);
#pragma unroll
    for (int t = 0; t < 4; ++t) acc[t] = wmma_bf16(aP, bP[t], acc[t]);

    // issue loads for k0+64 (buffer P) before consuming buffer Q
    if (k0 + 64 < K) {
      aP = load_frag_off(A + k0 + 64, offA);
#pragma unroll
      for (int t = 0; t < 4; ++t)
        bP[t] = load_frag_off(Bt + t * tstep + k0 + 64, offB);
    }
#pragma unroll
    for (int t = 0; t < 4; ++t) acc[t] = wmma_bf16(aQ, bQ[t], acc[t]);
  }

  // C/D layout: element (vgpr v, lane) -> row = v + 8*(lane>=16), col = lane&15
  const unsigned rbase = (unsigned)(m0 + 8 * (lane >> 4));
  const unsigned cbase = (unsigned)(n0 + (lane & 15));
#pragma unroll
  for (int t = 0; t < 4; ++t)
#pragma unroll
    for (int v = 0; v < 8; ++v) {
      unsigned idx = (rbase + v) * (unsigned)N + cbase + t * 16;
      if (Cb) Cb[idx] = (__bf16)acc[t][v];
      else    Cf[idx] = acc[t][v];
    }
}

// ---- RMS-norm + RoPE + head scatter ----------------------------------------
// One wave handles one (b, s, head-row) of 256 elements (8 per lane).
// h in [0,16): Q head -> Qo[B][NH][S][HD]
// h in [16,24): K head -> Ko[B][NKV][S][HD]
// h in [24,32): V head -> Vto[B][NKV][HD][S]   (transposed for PV B-frags)

__global__ void norm_rope_scatter(const __bf16* __restrict__ qkv,
                                  const float* __restrict__ cosp,
                                  const float* __restrict__ sinp,
                                  const float* __restrict__ qw,
                                  const float* __restrict__ kw,
                                  __bf16* __restrict__ Qo,
                                  __bf16* __restrict__ Ko,
                                  __bf16* __restrict__ Vto) {
  const int lane = threadIdx.x & 31;
  const int wave = threadIdx.x >> 5;
  const int gid  = blockIdx.x * 8 + wave;
  const int b    = gid >> 16;          // / (S_*32)
  const int rem  = gid & 0xFFFF;
  const int s    = rem >> 5;
  const int h    = rem & 31;
  const __bf16* row = qkv + (size_t)(b * S_ + s) * QKVF_;
  const int d0 = lane * 8;

  if (h >= 24) {                                    // V pass-through, transpose
    const __bf16* src = row + 6144 + (h - 24) * HD_;
    __bf16* dst = Vto + ((size_t)(b * NKV_ + (h - 24)) * HD_) * S_;
#pragma unroll
    for (int i = 0; i < 8; ++i) dst[(size_t)(d0 + i) * S_ + s] = src[d0 + i];
    return;
  }

  const __bf16* src = (h < 16) ? (row + h * HD_)
                               : (row + NH_ * HD_ + (h - 16) * HD_);
  float x[8], p[8];
#pragma unroll
  for (int i = 0; i < 8; ++i) x[i] = (float)src[d0 + i];
  const int pd0 = d0 ^ 128;                          // rotate_half partner
#pragma unroll
  for (int i = 0; i < 8; ++i) p[i] = (float)src[pd0 + i];

  float ss = 0.f;
#pragma unroll
  for (int i = 0; i < 8; ++i) ss += x[i] * x[i];
#pragma unroll
  for (int off = 1; off < 32; off <<= 1) ss += __shfl_xor(ss, off, 32);
  const float inv = rsqrtf(ss * (1.0f / 256.0f) + EPS_);

  const float* nw = (h < 16) ? qw : kw;
  const float* cs = cosp + (size_t)(b * S_ + s) * HD_;
  const float* sn = sinp + (size_t)(b * S_ + s) * HD_;
  __bf16* dst = (h < 16)
      ? Qo + ((size_t)(b * NH_ + h) * S_ + s) * HD_
      : Ko + ((size_t)(b * NKV_ + (h - 16)) * S_ + s) * HD_;
#pragma unroll
  for (int i = 0; i < 8; ++i) {
    const int d = d0 + i;
    const float xn  = x[i] * inv * (1.f + nw[d]);
    const float pn  = p[i] * inv * (1.f + nw[pd0 + i]);
    const float rot = (d < 128) ? -pn : pn;          // [-x2, x1]
    dst[d] = (__bf16)(xn * cs[d] + rot * sn[d]);
  }
}

// ---- flash attention (non-causal, GQA n_rep=2) -----------------------------
// Block = 128 threads = 4 waves; each wave owns 16 query rows; online softmax
// over 64-key blocks.  P is restaged bf16 through LDS to form A fragments.
// Uniform base + invariant lane offsets; lookahead-1 pipelining on Q*K^T and
// P*V.  launch_bounds(128,1): ~240 VGPRs of live state (O accumulators!) must
// stay in registers.

__global__ void __launch_bounds__(128, 1)
attn_kernel(const __bf16* __restrict__ Q,
            const __bf16* __restrict__ Km,
            const __bf16* __restrict__ Vt,
            __bf16* __restrict__ Out) {
  __shared__ __align__(16) __bf16 plds[4][16][64];
  const int lane = threadIdx.x & 31;
  const int wave = threadIdx.x >> 5;
  const int bh = blockIdx.x >> 5;              // / (S/64)
  const int qb = blockIdx.x & 31;
  const int b  = bh >> 4;
  const int h  = bh & 15;
  const int hkv = h >> 1;                      // n_rep = 2
  const __bf16* Qb = Q  + ((size_t)(b * NH_  + h)   * S_)  * HD_;
  const __bf16* Kb = Km + ((size_t)(b * NKV_ + hkv) * S_)  * HD_;
  const __bf16* Vb = Vt + ((size_t)(b * NKV_ + hkv) * HD_) * S_;
  const int q0 = qb * 64 + wave * 16;

  // invariant per-lane element offsets
  const unsigned offQ = frag_off(q0, HD_, lane);   // Q rows fixed at q0
  const unsigned offK = frag_off(0, HD_, lane);    // K row base added uniformly
  const unsigned offV = frag_off(0, S_, lane);     // V row base added uniformly

  v8f O[16] = {};
  v8f m, l;
#pragma unroll
  for (int v = 0; v < 8; ++v) { m[v] = -1e30f; l[v] = 0.f; }

  for (int kb = 0; kb < S_; kb += 64) {
    const __bf16* Kblk = Kb + (unsigned)kb * HD_;
    // ---- S = (Q K^T) * scale : 4 tiles of 16x16, reduce over d=256.
    // Flattened (d-step, tile) loop, lookahead-1 on both fragments.
    v8f Sacc[4] = {};
    v16bf aq   = load_frag_off(Qb, offQ);
    v16bf kcur = load_frag_off(Kblk, offK);
    v16bf anext = aq, knext = kcur;
#pragma unroll
    for (int i = 0; i < 32; ++i) {              // i = d_step*4 + tile
      const int t = i & 3;
      if (i < 31) {
        const int ni = i + 1;
        knext = load_frag_off(Kblk + (ni & 3) * (16 * HD_) + (ni >> 2) * 32, offK);
        if ((ni & 3) == 0)
          anext = load_frag_off(Qb + (ni >> 2) * 32, offQ);
      }
      Sacc[t] = wmma_bf16(aq, kcur, Sacc[t]);
      kcur = knext;
      if (t == 3) aq = anext;
    }
#pragma unroll
    for (int t = 0; t < 4; ++t)
#pragma unroll
      for (int v = 0; v < 8; ++v) Sacc[t][v] *= SCALE_;

    // ---- online softmax (rows live across 16-lane halves)
    v8f pm;
#pragma unroll
    for (int v = 0; v < 8; ++v) {
      float mx = Sacc[0][v];
#pragma unroll
      for (int t = 1; t < 4; ++t) mx = fmaxf(mx, Sacc[t][v]);
      pm[v] = mx;
    }
#pragma unroll
    for (int off = 1; off < 16; off <<= 1)
#pragma unroll
      for (int v = 0; v < 8; ++v)
        pm[v] = fmaxf(pm[v], __shfl_xor(pm[v], off, 32));

    v8f sc;
#pragma unroll
    for (int v = 0; v < 8; ++v) {
      float mn = fmaxf(m[v], pm[v]);
      sc[v] = __expf(m[v] - mn);
      m[v] = mn;
    }
    v8f rs = {};
#pragma unroll
    for (int t = 0; t < 4; ++t)
#pragma unroll
      for (int v = 0; v < 8; ++v) {
        float pe = __expf(Sacc[t][v] - m[v]);
        Sacc[t][v] = pe;
        rs[v] += pe;
      }
#pragma unroll
    for (int off = 1; off < 16; off <<= 1)
#pragma unroll
      for (int v = 0; v < 8; ++v) rs[v] += __shfl_xor(rs[v], off, 32);
#pragma unroll
    for (int v = 0; v < 8; ++v) l[v] = l[v] * sc[v] + rs[v];
#pragma unroll
    for (int i = 0; i < 16; ++i)
#pragma unroll
      for (int v = 0; v < 8; ++v) O[i][v] *= sc[v];

    // ---- stage P (bf16) into LDS as row-major [16 rows][64 keys]
    const int rr = 8 * (lane >> 4);
    const int cc = lane & 15;
#pragma unroll
    for (int t = 0; t < 4; ++t)
#pragma unroll
      for (int v = 0; v < 8; ++v)
        plds[wave][rr + v][t * 16 + cc] = (__bf16)Sacc[t][v];
    __syncthreads();

    // ---- O += P V : B fragment from transposed V (Vt[d][key], contiguous),
    // lookahead-1 on the V fragment.
#pragma unroll
    for (int kc = 0; kc < 2; ++kc) {
      v16bf a = load_frag_lds(&plds[wave][0][0], 64, kc * 32, lane);
      const __bf16* Vblk = Vb + kb + kc * 32;
      v16bf vcur = load_frag_off(Vblk, offV);
      v16bf vnext = vcur;
#pragma unroll
      for (int td = 0; td < 16; ++td) {
        if (td < 15)
          vnext = load_frag_off(Vblk + (td + 1) * (16 * S_), offV);
        O[td] = wmma_bf16(a, vcur, O[td]);
        vcur = vnext;
      }
    }
    __syncthreads();
  }

  // ---- normalize and scatter to attn buffer [B*S][NH*HD] (bf16)
  const unsigned rbase = (unsigned)((size_t)b * S_ + q0 + 8 * (lane >> 4));
  const unsigned cb = (unsigned)(h * HD_ + (lane & 15));
#pragma unroll
  for (int td = 0; td < 16; ++td)
#pragma unroll
    for (int v = 0; v < 8; ++v) {
      const float o = O[td][v] / l[v];
      Out[(size_t)(rbase + v) * (NH_ * HD_) + cb + td * 16] = (__bf16)o;
    }
}

// ---------------------------------------------------------------------------

extern "C" void kernel_launch(void* const* d_in, const int* in_sizes, int n_in,
                              void* d_out, int out_size, void* d_ws, size_t ws_size,
                              hipStream_t stream) {
  const float* hidden = (const float*)d_in[0];
  const float* cosp   = (const float*)d_in[1];
  const float* sinp   = (const float*)d_in[2];
  const float* wqkv   = (const float*)d_in[3];
  const float* wo     = (const float*)d_in[4];
  const float* qw     = (const float*)d_in[5];
  const float* kw     = (const float*)d_in[6];
  float* out = (float*)d_out;

  char* ws = (char*)d_ws;
  // workspace layout (bytes)
  __bf16* hid_b  = (__bf16*)(ws);                       //  16 MiB: [4096][2048]
  __bf16* wqkv_t = (__bf16*)(ws + 16777216ull);         //  32 MiB: [8192][2048]
  __bf16* wo_t   = (__bf16*)(ws + 50331648ull);         //  16 MiB: [2048][4096]
  __bf16* qkv_b  = (__bf16*)(ws + 67108864ull);         //  64 MiB: [4096][8192]
  __bf16* Qb     = (__bf16*)(ws + 134217728ull);        //  32 MiB: [2][16][2048][256]
  __bf16* Kb     = (__bf16*)(ws + 167772160ull);        //  16 MiB: [2][8][2048][256]
  __bf16* Vtb    = (__bf16*)(ws + 184549376ull);        //  16 MiB: [2][8][256][2048]
  __bf16* attn_b = (__bf16*)(ws + 201326592ull);        //  32 MiB: [4096][4096]

  // 1) operand conversion / transposition
  cvt_f32_to_bf16<<<32768, 256, 0, stream>>>(hidden, hid_b, B_ * S_ * HID_);
  transpose_f32_to_bf16<<<65536, 256, 0, stream>>>(wqkv, wqkv_t, HID_, QKVF_);
  transpose_f32_to_bf16<<<32768, 256, 0, stream>>>(wo, wo_t, NH_ * HD_, HID_);

  // 2) QKV projection: [4096,2048] x [2048,8192] -> bf16 [4096,8192]
  //    waves = (4096/16)*(8192/64) = 32768 -> 4096 blocks of 8 waves
  gemm16x64_bf16<<<4096, 256, 0, stream>>>(hid_b, wqkv_t, qkv_b, nullptr,
                                           B_ * S_, QKVF_, HID_);

  // 3) RMS-norm + RoPE + scatter: B*S*32 head-rows, 1 wave each
  norm_rope_scatter<<<16384, 256, 0, stream>>>(qkv_b, cosp, sinp, qw, kw,
                                               Qb, Kb, Vtb);

  // 4) attention: grid = B*NH*(S/64) = 1024, 4 waves/block
  attn_kernel<<<1024, 128, 0, stream>>>(Qb, Kb, Vtb, attn_b);

  // 5) output projection: [4096,4096] x [4096,2048] -> fp32 d_out
  //    waves = 256*32 = 8192 -> 1024 blocks
  gemm16x64_bf16<<<1024, 256, 0, stream>>>(attn_b, wo_t, nullptr, out,
                                           B_ * S_, HID_, NH_ * HD_);
}
